// PairEnergyHead_80719615361638
// MI455X (gfx1250) — compile-verified, matrix operand order checked by reading, not run on hardware
//
#include <hip/hip_runtime.h>
#include <hip/hip_bf16.h>
#include <math.h>

typedef __attribute__((ext_vector_type(16))) _Float16 v16h;
typedef __attribute__((ext_vector_type(8)))  float    v8f;

#define L   256
#define DH  128
#define ZD  16
#define PD  120
#define HD  256
// W1 row offsets inside the 650-row weight
#define OFF_HI    0
#define OFF_HJ    128
#define OFF_ZI    256
#define OFF_ZJ    272
#define OFF_PI    288
#define OFF_PJ    408
#define OFF_PIJ   528
#define OFF_DIST  648
#define OFF_CANON 649

// compile-time triu(16,1) index tables (row-major order, matches np.triu_indices)
struct Tri {
  int a[128]; int b[128];
  constexpr Tri() : a(), b() {
    int t = 0;
    for (int i = 0; i < 16; ++i)
      for (int j = i + 1; j < 16; ++j) { a[t] = i; b[t] = j; ++t; }
    for (; t < 128; ++t) { a[t] = 0; b[t] = 0; }
  }
};
__device__ constexpr Tri TRI{};

// ---------------------------------------------------------------------------
// Kernel 1: per-token precompute  A[b,i,:] (i-side rows of W1 + bias) and
//           Bv[b,j,:] (j-side rows).  264x256 dot per token -> negligible cost.
// ---------------------------------------------------------------------------
__global__ __launch_bounds__(256) void k_precompute(
    const float* __restrict__ h, const float* __restrict__ z, const float* __restrict__ p,
    const float* __restrict__ W1, const float* __restrict__ b1,
    float* __restrict__ Aw, float* __restrict__ Bw)
{
  int bi = blockIdx.x;          // b*L + i
  int n  = threadIdx.x;         // hidden channel 0..255
  const float* hp = h + bi * DH;
  const float* zp = z + bi * ZD;
  const float* pp = p + bi * PD;
  float av = b1[n], bv = 0.f;
  #pragma unroll 4
  for (int k = 0; k < DH; ++k) {
    float x = hp[k];
    av = fmaf(x, W1[(OFF_HI + k) * HD + n], av);
    bv = fmaf(x, W1[(OFF_HJ + k) * HD + n], bv);
  }
  #pragma unroll
  for (int k = 0; k < ZD; ++k) {
    float x = zp[k];
    av = fmaf(x, W1[(OFF_ZI + k) * HD + n], av);
    bv = fmaf(x, W1[(OFF_ZJ + k) * HD + n], bv);
  }
  #pragma unroll 4
  for (int k = 0; k < PD; ++k) {
    float x = pp[k];
    av = fmaf(x, W1[(OFF_PI + k) * HD + n], av);
    bv = fmaf(x, W1[(OFF_PJ + k) * HD + n], bv);
  }
  Aw[bi * HD + n] = av;
  Bw[bi * HD + n] = bv;
}

// ---------------------------------------------------------------------------
// Kernel 1b: pack W1[528:648] (plucker rows, padded K=120->128) into f16
// pre-swizzled B-operand layout: [(s*16+ng)*32 + lane]*8 + r dwords, where
// lane<16: r<4 -> k=2r,2r+1 ; r>=4 -> k=2r+8,...  lane>=16: +8 / +16.
// ---------------------------------------------------------------------------
__global__ __launch_bounds__(256) void k_packW(const float* __restrict__ W1,
                                               unsigned* __restrict__ Wp)
{
  int t    = blockIdx.x * 256 + threadIdx.x;   // < 16384
  int r    = t & 7;
  int lane = (t >> 3) & 31;
  int ng   = (t >> 8) & 15;
  int s    = t >> 12;
  int hi   = lane >> 4;
  int n    = ng * 16 + (lane & 15);
  int kl   = 2 * r + (r < 4 ? (hi ? 8 : 0) : (hi ? 16 : 8));
  int k0   = 32 * s + kl, k1 = k0 + 1;
  float f0 = (k0 < PD) ? W1[(OFF_PIJ + k0) * HD + n] : 0.f;
  float f1 = (k1 < PD) ? W1[(OFF_PIJ + k1) * HD + n] : 0.f;
  union { _Float16 e[2]; unsigned u; } pk;
  pk.e[0] = (_Float16)f0; pk.e[1] = (_Float16)f1;
  Wp[t] = pk.u;
}

// ---------------------------------------------------------------------------
// Kernel 2: pairwise core.  One wave per 16-pair M-tile (fixed b,i; 16 j's).
// The 64 KB packed-weight tensor is staged once per block into LDS with
// GLOBAL_LOAD_ASYNC_TO_LDS_B128 (ASYNCcnt path); all four B operands of an
// N-group are ds-loaded into distinct registers BEFORE the WMMA chain so the
// LDS returns overlap the WMMAs (staggered dscnt waits instead of wait-0).
// hidden = WMMA(p_ij, Wp) + A[b,i] + Bv[b,j] + dist*wd + canon*wc ; GELU ;
// dot with W2 via shfl butterfly -> unsymmetrized E[b,i,j].
// ---------------------------------------------------------------------------
__global__ __launch_bounds__(256) void k_pair(
    const float* __restrict__ z, const unsigned char* __restrict__ canon,
    const float* __restrict__ W1, const float* __restrict__ W2,
    const float* __restrict__ b2,
    const float* __restrict__ Aw, const float* __restrict__ Bw,
    const unsigned* __restrict__ Wp, float* __restrict__ E)
{
  __shared__ __align__(16) unsigned ldsWp[16384];   // 64 KB packed B operands

  // ---- async stage Wp -> LDS: 256 thr x 16 B x 16 iters = 64 KB.
  // INST_OFFSET is added to BOTH the global and LDS addresses (ISA 8 §4.4),
  // so one base address pair + 16 immediate offsets covers it.
  {
    unsigned stageOff = (unsigned)threadIdx.x * 16u;
    unsigned ldsAddr  = (unsigned)(uintptr_t)(void*)ldsWp + stageOff;
    #pragma unroll
    for (int it = 0; it < 16; ++it) {
      asm volatile("global_load_async_to_lds_b128 %0, %1, %2 offset:%3"
                   :
                   : "v"(ldsAddr), "v"(stageOff), "s"(Wp), "n"(it * 4096)
                   : "memory");
    }
    asm volatile("s_wait_asynccnt 0" ::: "memory");
  }
  __syncthreads();

  int tile = blockIdx.x;              // b*(L*2) + i*2 + jblk
  int jblk = tile & 1;
  int i    = (tile >> 1) & (L - 1);
  int b    = tile >> 9;
  int lane = threadIdx.x & 31;
  int wave = threadIdx.x >> 5;
  int hi   = lane >> 4;
  int m    = lane & 15;               // pair row within tile (both half-waves)
  int j0   = jblk * 128 + wave * 16;
  int bi   = b * L + i;

  // z_i (shared across wave) and z_j for this lane's row
  float zi[16], zj[16];
  const float* zip = z + bi * ZD;
  const float* zjp = z + (b * L + j0 + m) * ZD;
  #pragma unroll
  for (int k = 0; k < 16; ++k) { zi[k] = zip[k]; zj[k] = zjp[k]; }

  // pass 1: plucker sum of squares (fully unrolled, constant register indices)
  float ss = 0.f;
  #pragma unroll
  for (int a = 0; a < 15; ++a) {
    #pragma unroll
    for (int c2 = a + 1; c2 < 16; ++c2) {
      float v = zi[a] * zj[c2] - zi[c2] * zj[a];
      ss = fmaf(v, v, ss);
    }
  }
  float rn = 1.0f / fmaxf(sqrtf(ss), 1e-8f);

  // pass 2: build the four 16x32 f16 A-operands in WMMA register layout
  v16h aop[4];
  #pragma unroll
  for (int s = 0; s < 4; ++s) {
    union { v16h v; _Float16 e[16]; } av;
    #pragma unroll
    for (int e = 0; e < 16; ++e) {
      int r = e >> 1, odd = e & 1;
      int kA = 32 * s + 2 * r + (r < 4 ? 0 : 8) + odd;    // lanes 0-15
      int kB = 32 * s + 2 * r + (r < 4 ? 8 : 16) + odd;   // lanes 16-31
      float v0 = (kA < PD)
          ? (zi[TRI.a[kA]] * zj[TRI.b[kA]] - zi[TRI.b[kA]] * zj[TRI.a[kA]]) * rn : 0.f;
      float v1 = (kB < PD)
          ? (zi[TRI.a[kB]] * zj[TRI.b[kB]] - zi[TRI.b[kB]] * zj[TRI.a[kB]]) * rn : 0.f;
      av.e[e] = (_Float16)(hi ? v1 : v0);
    }
    aop[s] = av.v;
  }

  float acc[8];
  #pragma unroll
  for (int r = 0; r < 8; ++r) acc[r] = 0.f;

  int   nl    = lane & 15;
  float distc = 1.0f / (float)L;

  for (int ng = 0; ng < 16; ++ng) {
    // load ALL four B operands into distinct registers first -> one 8-deep
    // ds_load_b128 clause, then WMMAs retire against staggered dscnt waits.
    v16h bm[4];
    #pragma unroll
    for (int s = 0; s < 4; ++s) {
      const unsigned* bp = ldsWp + (ng * 32 + lane) * 8 + s * 4096;
      bm[s] = *(const v16h*)bp;
    }
    v8f c = {};
    #pragma unroll
    for (int s = 0; s < 4; ++s) {
      c = __builtin_amdgcn_wmma_f32_16x16x32_f16(
              false, aop[s], false, bm[s], (short)0, c, false, false);
    }
    int   n     = ng * 16 + nl;
    float a_val = Aw[bi * HD + n];
    float wd    = W1[OFF_DIST  * HD + n];
    float wc    = W1[OFF_CANON * HD + n];
    float w2    = W2[n];
    #pragma unroll
    for (int r = 0; r < 8; ++r) {
      int row = r + 8 * hi;                     // D layout: VGPR r -> M=r / r+8
      int j   = j0 + row;
      float x = c[r] + a_val + Bw[(b * L + j) * HD + n];
      x = fmaf(fabsf((float)(i - j)) * distc, wd, x);
      x = fmaf((float)canon[bi * L + j], wc, x);
      float g = 0.5f * x * (1.0f + erff(x * 0.70710678118654752f));  // exact GELU
      acc[r] = fmaf(g, w2, acc[r]);
    }
  }

  // reduce the 16 N-lanes of each half-wave; lanes 0/16 write 8 rows each
  float b2v = b2[0];
  #pragma unroll
  for (int r = 0; r < 8; ++r) {
    float v = acc[r];
    v += __shfl_xor(v, 1, 16);
    v += __shfl_xor(v, 2, 16);
    v += __shfl_xor(v, 4, 16);
    v += __shfl_xor(v, 8, 16);
    acc[r] = v;
  }
  if (nl == 0) {
    #pragma unroll
    for (int r = 0; r < 8; ++r)
      E[bi * L + j0 + r + 8 * hi] = acc[r] + b2v;
  }
}

// ---------------------------------------------------------------------------
// Kernel 3: symmetrize + sequence mask.
// ---------------------------------------------------------------------------
__global__ __launch_bounds__(256) void k_sym(
    const float* __restrict__ E, const unsigned char* __restrict__ sm,
    float* __restrict__ out)
{
  int t = blockIdx.x * 256 + threadIdx.x;
  int j = t & (L - 1);
  int i = (t >> 8) & (L - 1);
  int b = t >> 16;
  bool valid = sm[b * L + i] && sm[b * L + j];
  float e = 0.5f * (E[t] + E[(b * L + j) * L + i]);
  out[t] = valid ? e : 30000.0f;
}

// ---------------------------------------------------------------------------
extern "C" void kernel_launch(void* const* d_in, const int* in_sizes, int n_in,
                              void* d_out, int out_size, void* d_ws, size_t ws_size,
                              hipStream_t stream)
{
  const float*         h  = (const float*)d_in[0];
  const float*         z  = (const float*)d_in[1];
  const float*         p  = (const float*)d_in[2];
  const unsigned char* sm = (const unsigned char*)d_in[3];
  const unsigned char* cm = (const unsigned char*)d_in[4];
  const float*         W1 = (const float*)d_in[5];
  const float*         b1 = (const float*)d_in[6];
  const float*         W2 = (const float*)d_in[7];
  const float*         b2 = (const float*)d_in[8];
  float*               out = (float*)d_out;

  const int B = 4;
  float*    Aw = (float*)d_ws;                  // B*L*HD floats (1 MB)
  float*    Bw = Aw + B * L * HD;               // B*L*HD floats (1 MB)
  unsigned* Wp = (unsigned*)(Bw + B * L * HD);  // 16384 dwords (64 KB)
  float*    E  = (float*)(Wp + 16384);          // B*L*L floats (1 MB)

  k_precompute<<<B * L,         256, 0, stream>>>(h, z, p, W1, b1, Aw, Bw);
  k_packW     <<<64,            256, 0, stream>>>(W1, Wp);
  k_pair      <<<B * L * 2,     256, 0, stream>>>(z, cm, W1, W2, b2, Aw, Bw, Wp, E);
  k_sym       <<<B * L * L/256, 256, 0, stream>>>(E, sm, out);
}
